// ForestRuthIntegrator_81466939671256
// MI455X (gfx1250) — matrix operand order repaired and checked by reading, not compiled, not run on hardware
//
#include <hip/hip_runtime.h>

typedef __attribute__((ext_vector_type(16))) _Float16 v16h;
typedef __attribute__((ext_vector_type(8)))  float    v8f;
typedef __attribute__((ext_vector_type(4)))  unsigned u32x4;
typedef __attribute__((ext_vector_type(8)))  int      i32x8;
typedef __attribute__((ext_vector_type(4)))  int      i32x4;

#define ND   16384
#define DD   1024
#define RR   256

#define BLK  1024          // 32 waves (wave32)
#define MROW 64            // rows of N per block
#define NBLK (ND / MROW)   // 256 blocks

#define SVH  1040          // v_h stride (f16 elems), even, padded
#define SH2  272           // h2 stride (f16 elems), even, padded

#define VH_BYTES  (MROW * SVH * 2)          // 133120
#define H2_BYTES  (MROW * SH2 * 2)          // 34816
#define UST_OFF   (VH_BYTES + H2_BYTES)     // 167936
#define UCHUNK_B  65536                     // 64 pair-rows x 256 dwords = K=128 f16
#define NCHUNK    8                         // 8 chunks * K128 = K1024
#define SMEM_B    (UST_OFF + 2 * UCHUNK_B)  // 299008 (< 320KB)

// Forest-Ruth: THETA = 1/(2 - 2^(1/3)), dt = 0.01
#define THETA_ 1.3512071919596578
#define DT_    0.01

// ---------------- helpers ----------------

__device__ __forceinline__ v8f wmma16(v16h a, v16h b, v8f c) {
  return __builtin_amdgcn_wmma_f32_16x16x32_f16(false, a, false, b, (short)0, c,
                                                false, false);
}

// A fragment (16x32 f16) from row-major LDS [16][stride]
__device__ __forceinline__ v16h load_a_frag(const _Float16* base, int stride,
                                            int kb, int lane) {
  const int r  = lane & 15;
  const int hi = (lane >> 4) << 4;             // 0 or 16
  const unsigned* p = (const unsigned*)(base + r * stride);
  union { unsigned u[8]; v16h h; } A;
#pragma unroll
  for (int i = 0; i < 8; ++i) {
    const int koff = i << 1;
    A.u[i] = p[(kb + hi + koff) >> 1];
  }
  return A.h;
}

// B fragment (32x16 f16) from packed global: P[(k/2)*ncols + n] = (f16(k), f16(k+1))
__device__ __forceinline__ v16h load_b_frag(const unsigned* __restrict__ P,
                                            int ncols, int kb, int nb, int lane) {
  const int n  = nb + (lane & 15);
  const int kh = (lane >> 4) << 3;
  const unsigned* p = P + (size_t)((kb >> 1) + kh) * ncols + n;
  union { unsigned u[8]; v16h h; } B;
#pragma unroll
  for (int i = 0; i < 8; ++i) B.u[i] = p[(size_t)i * ncols];
  return B.h;
}

// B fragment from TDM-staged LDS chunk laid out [64 pair-rows][256 dwords]
__device__ __forceinline__ v16h load_b_frag_lds(const unsigned* Pl, int kb_local,
                                                int nb, int lane) {
  const int n  = nb + (lane & 15);
  const int kh = (lane >> 4) << 3;
  const unsigned* p = Pl + ((kb_local >> 1) + kh) * 256 + n;
  union { unsigned u[8]; v16h h; } B;
#pragma unroll
  for (int i = 0; i < 8; ++i) B.u[i] = p[i * 256];
  return B.h;
}

__device__ __forceinline__ unsigned pack2(float a, float b) {
  _Float16 ha = (_Float16)a, hb = (_Float16)b;
  unsigned short ua, ub;
  __builtin_memcpy(&ua, &ha, 2);
  __builtin_memcpy(&ub, &hb, 2);
  return (unsigned)ua | ((unsigned)ub << 16);
}

// TDM: DMA one U chunk (64 pair-rows x 256 dwords = 64 KB) into LDS (ISA 8.3-8.5)
__device__ __forceinline__ void tdm_issue_u(const unsigned* __restrict__ U2,
                                            int chunk, unsigned lds_off) {
  const unsigned long long ga =
      (unsigned long long)(uintptr_t)U2 + (unsigned long long)chunk * UCHUNK_B;
  u32x4 g0;
  g0[0] = 1u;                                            // count=1, normal mode
  g0[1] = lds_off;                                       // lds_addr (bytes)
  g0[2] = (unsigned)ga;                                  // global_addr[31:0]
  g0[3] = (unsigned)((ga >> 32) & 0x1FFFFFFu) | (2u << 30); // addr[56:32] | type=2
  i32x8 g1;
  g1[0] = (int)(2u << 16);       // data_size=2 (4-byte elems), wg_mask=0
  g1[1] = (int)(256u << 16);     // tensor_dim0 = 256 dwords
  g1[2] = (int)(512u << 16);     // tensor_dim1 = 512 pair-rows
  g1[3] = (int)(256u << 16);     // tile_dim0 = 256
  g1[4] = 64;                    // tile_dim1 = 64 pair-rows (K=128 f16)
  g1[5] = 256;                   // tensor_dim0_stride = 256
  g1[6] = 0;
  g1[7] = 0;
  i32x4 z4 = {0, 0, 0, 0};
  i32x8 z8 = {0, 0, 0, 0, 0, 0, 0, 0};
  __builtin_amdgcn_tensor_load_to_lds(g0, g1, z4, z4, z8, 0);
}

// ---------------- prologue: pack U, W to f16-pair layout ----------------

__global__ void fr_pack_kernel(const float* __restrict__ U, const float* __restrict__ W,
                               unsigned* __restrict__ U2, unsigned* __restrict__ W2) {
  const int i = blockIdx.x * blockDim.x + threadIdx.x;   // 0..131071 (both arrays)
  {
    const int kp = i >> 8, n = i & 255;                  // U: [1024][256]
    U2[i] = pack2(U[(2 * kp) * RR + n], U[(2 * kp + 1) * RR + n]);
  }
  {
    const int kp = i >> 10, n = i & 1023;                // W: [256][1024]
    W2[i] = pack2(W[(2 * kp) * DD + n], W[(2 * kp + 1) * DD + n]);
  }
}

// ---------------- fused Forest-Ruth integrator ----------------

__global__ __launch_bounds__(BLK)
void fr_step_kernel(const float* __restrict__ x_in, const float* __restrict__ v_in,
                    const float* __restrict__ f_in,
                    const unsigned* __restrict__ U2, const unsigned* __restrict__ W2,
                    const int* __restrict__ steps_p,
                    float* __restrict__ x_out, float* __restrict__ v_out) {
  extern __shared__ char smem[];
  _Float16* v_h = (_Float16*)smem;                 // [64][SVH] f16 of v
  _Float16* h2s = (_Float16*)(smem + VH_BYTES);    // [64][SH2] f16 of (v@U)^2

  const int tid  = threadIdx.x;
  const int lane = tid & 31;
  const int wave = tid >> 5;
  const int cL   = lane & 15;
  const int rb   = (lane >> 4) << 3;               // 0 or 8
  const int rowBase = blockIdx.x * MROW;

  // v/dx ownership == GEMM2 output fragments:
  //   (q,m,i): row = 16m + rb + i, col = 32*wave + 16*q + cL
  float vv[2][4][8], dxx[2][4][8];
#pragma unroll
  for (int q = 0; q < 2; ++q) {
    const int col = 32 * wave + 16 * q + cL;
#pragma unroll
    for (int m = 0; m < 4; ++m)
#pragma unroll
      for (int i = 0; i < 8; ++i) {
        const int row = 16 * m + rb + i;
        const float t = v_in[(size_t)(rowBase + row) * DD + col];
        vv[q][m][i]  = t;
        dxx[q][m][i] = 0.0f;
        v_h[row * SVH + col] = (_Float16)t;
      }
  }
  __syncthreads();

  const int steps = steps_p[0];

  const float cdt[4] = {(float)(0.5 * THETA_ * DT_), (float)(0.5 * (1.0 - THETA_) * DT_),
                        (float)(0.5 * (1.0 - THETA_) * DT_), (float)(0.5 * THETA_ * DT_)};
  const float ddt[4] = {(float)(THETA_ * DT_), (float)((1.0 - 2.0 * THETA_) * DT_),
                        (float)(THETA_ * DT_), (float)(THETA_ * DT_)};

  const int nR = wave & 15;               // GEMM1 n-tile (R = 16 tiles)
  const int mb = (wave >> 4) << 1;        // GEMM1 m-subtiles {mb, mb+1}
  const float* __restrict__ fbase = f_in + (size_t)rowBase * DD;

#pragma unroll 1
  for (int s = 0; s < steps; ++s) {
#pragma unroll 1
    for (int sub = 0; sub < 4; ++sub) {
      const float cc = cdt[sub];
      const float dd = ddt[sub];

      // dx += c*dt*v (uses pre-update v; single wrap deferred to the end)
#pragma unroll
      for (int q = 0; q < 2; ++q)
#pragma unroll
        for (int m = 0; m < 4; ++m)
#pragma unroll
          for (int i = 0; i < 8; ++i) dxx[q][m][i] += cc * vv[q][m][i];

      // ---- GEMM1: h = v @ U, U double-buffered via TDM DMA ----
      {
        v8f acc0 = {}, acc1 = {};
        if (wave == 31)
          tdm_issue_u(U2, 0, (unsigned)(uintptr_t)(smem + UST_OFF));
#pragma unroll 1
        for (int c = 0; c < NCHUNK; ++c) {
          if (wave == 31) __builtin_amdgcn_s_wait_tensorcnt(0);
          __syncthreads();                       // chunk c data visible to all
          if (wave == 31 && c + 1 < NCHUNK)
            tdm_issue_u(U2, c + 1,
                        (unsigned)(uintptr_t)(smem + UST_OFF + ((c + 1) & 1) * UCHUNK_B));
          const unsigned* Bst = (const unsigned*)(smem + UST_OFF + (c & 1) * UCHUNK_B);
          const int kg = c * 128;
#pragma unroll
          for (int kk = 0; kk < 128; kk += 32) {
            v16h b  = load_b_frag_lds(Bst, kk, nR * 16, lane);
            v16h a0 = load_a_frag(v_h + (16 * mb) * SVH, SVH, kg + kk, lane);
            v16h a1 = load_a_frag(v_h + (16 * (mb + 1)) * SVH, SVH, kg + kk, lane);
            acc0 = wmma16(a0, b, acc0);
            acc1 = wmma16(a1, b, acc1);
          }
          __syncthreads();                       // done reading buf (c&1)
        }
        const int col = nR * 16 + cL;
#pragma unroll
        for (int i = 0; i < 8; ++i) {
          const float h0 = acc0[i], h1 = acc1[i];
          h2s[(16 * mb + rb + i) * SH2 + col]       = (_Float16)(h0 * h0);
          h2s[(16 * (mb + 1) + rb + i) * SH2 + col] = (_Float16)(h1 * h1);
        }
      }
      __syncthreads();                           // h2 ready

      // ---- GEMM2: t = h2 @ W (wave-exclusive W columns) + fused v-update ----
#pragma unroll 1
      for (int q = 0; q < 2; ++q) {
        const int nb = (2 * wave + q) * 16;
        v8f am[4] = {v8f{}, v8f{}, v8f{}, v8f{}};
#pragma unroll
        for (int kb = 0; kb < RR; kb += 32) {
          v16h b = load_b_frag(W2, DD, kb, nb, lane);
#pragma unroll
          for (int m = 0; m < 4; ++m) {
            v16h a = load_a_frag(h2s + (16 * m) * SH2, SH2, kb, lane);
            am[m] = wmma16(a, b, am[m]);
          }
        }
        const int col = nb + cL;
#pragma unroll
        for (int m = 0; m < 4; ++m)
#pragma unroll
          for (int i = 0; i < 8; ++i) {
            const float f  = fbase[(size_t)(16 * m + rb + i) * DD + col];
            const float nv = vv[q][m][i] + dd * (f - am[m][i]);
            vv[q][m][i] = nv;
            v_h[(16 * m + rb + i) * SVH + col] = (_Float16)nv;
          }
      }
      __syncthreads();                           // v_h ready for next GEMM1
    }
  }

  // Final: x_out = wrap(x_in + dx), v_out = v
  const float PI_F    = 3.14159265358979323846f;
  const float TWO_PI  = 6.28318530717958647692f;
  const float INV_2PI = 0.15915494309189533577f;
#pragma unroll
  for (int q = 0; q < 2; ++q) {
    const int col = 32 * wave + 16 * q + cL;
#pragma unroll
    for (int m = 0; m < 4; ++m)
#pragma unroll
      for (int i = 0; i < 8; ++i) {
        const size_t idx = (size_t)(rowBase + 16 * m + rb + i) * DD + col;
        float t = x_in[idx] + dxx[q][m][i] + PI_F;
        t -= floorf(t * INV_2PI) * TWO_PI;
        x_out[idx] = t - PI_F;
        v_out[idx] = vv[q][m][i];
      }
  }
}

// ---------------- launch ----------------

extern "C" void kernel_launch(void* const* d_in, const int* in_sizes, int n_in,
                              void* d_out, int out_size, void* d_ws, size_t ws_size,
                              hipStream_t stream) {
  (void)in_sizes; (void)n_in; (void)out_size; (void)ws_size;

  const float* x = (const float*)d_in[0];
  const float* v = (const float*)d_in[1];
  const float* f = (const float*)d_in[2];
  const float* U = (const float*)d_in[3];
  const float* W = (const float*)d_in[4];
  const int* steps = (const int*)d_in[5];

  float* x_out = (float*)d_out;
  float* v_out = x_out + (size_t)ND * DD;

  unsigned* U2 = (unsigned*)d_ws;                 // 512 KB
  unsigned* W2 = U2 + (DD / 2) * RR;              // 512 KB

  fr_pack_kernel<<<512, 256, 0, stream>>>(U, W, U2, W2);

  (void)hipFuncSetAttribute((const void*)fr_step_kernel,
                            hipFuncAttributeMaxDynamicSharedMemorySize, (int)SMEM_B);
  fr_step_kernel<<<NBLK, BLK, SMEM_B, stream>>>(x, v, f, U2, W2, steps, x_out, v_out);
}
